// LSTM_326417515128
// MI455X (gfx1250) — matrix operand order repaired and checked by reading, not compiled, not run on hardware
//
#include <hip/hip_runtime.h>
#include <hip/hip_bf16.h>

// ---------------------------------------------------------------------------
// LSTM on MI455X (gfx1250):
//   * bf16 WMMA (v_wmma_f32_16x16x32_bf16), fp32 accumulate
//   * weights pre-transposed to [col][gate][K] bf16: one lane's 4 gate
//     B-fragments span 12KB -> ALL B loads are immediate-offset b128 loads
//     off a single VGPR base (fits 24-bit signed offset); one pointer add
//     per 64-K iteration. W = 12.6 MB bf16, resident in 192MB L2.
//   * each wave owns 2 M-tiles x 4 gates for one 16-col slice: every
//     B-fragment feeds TWO WMMAs (32 FLOP per L2 byte)
//   * 2-slot pipeline on B fragments (L2 latency); A fragments just-in-time
//     from LDS via immediate-offset ds_load_b128 off one address register.
//   * grid = (2 batch tiles of 32) x (16 hidden slices of 64) = 32 WGs;
//     per-step sync scoped to the 16 WGs of one batch tile
//   * cell state c lives in registers for all 512 steps
// ---------------------------------------------------------------------------

#define B_SZ 64
#define T_SZ 512
#define D_SZ 512
#define H_SZ 1024
#define K_SZ 1536   // D_SZ + H_SZ
#define G4   4

typedef __attribute__((ext_vector_type(16))) __bf16 v16bf;
typedef __attribute__((ext_vector_type(8)))  float  v8f;

// workspace layout (bytes)
#define OFF_CNT  0                                   // u32 barrier counters
#define OFF_HBUF 256                                 // double-buffered h, bf16
#define SZ_HBUF  (2 * B_SZ * H_SZ * 2)               // 262144
#define OFF_XB   (OFF_HBUF + SZ_HBUF)                // x in bf16, [B][T][D]
#define SZ_XB    ((size_t)B_SZ * T_SZ * D_SZ * 2)    // 32 MB
#define OFF_WT   (OFF_XB + SZ_XB)                    // W^T bf16 [col][gate][K]
#define SZ_WT    ((size_t)G4 * H_SZ * K_SZ * 2)      // 12 MB

__device__ __forceinline__ float sigm_f(float x) {
  return 1.0f / (1.0f + __expf(-x));
}
__device__ __forceinline__ float tanh_f(float x) {
  return 2.0f * sigm_f(2.0f * x) - 1.0f;
}

// ---------------------------------------------------------------------------
// Init: zero counters + h buffers, convert x -> bf16, transpose W -> bf16
// Wt layout: [col][gate][K]  (col-major over the 4 gates' K-vectors)
// ---------------------------------------------------------------------------
__global__ void lstm_init_kernel(const float* __restrict__ x,
                                 const float* __restrict__ Wc,
                                 const float* __restrict__ Wi,
                                 const float* __restrict__ Wf,
                                 const float* __restrict__ Wo,
                                 unsigned char* __restrict__ ws) {
  unsigned*        zero32 = (unsigned*)(ws + OFF_CNT);
  __hip_bfloat16*  Xb     = (__hip_bfloat16*)(ws + OFF_XB);
  __hip_bfloat16*  Wt     = (__hip_bfloat16*)(ws + OFF_WT);
  const float* Wg[4] = {Wc, Wi, Wf, Wo};

  const size_t nZero = (size_t)(OFF_XB) / 4;             // counters + hbuf
  const size_t nX    = (size_t)B_SZ * T_SZ * D_SZ;
  const size_t nW    = (size_t)G4 * H_SZ * K_SZ;
  const size_t total = nZero + nX + nW;

  for (size_t i = (size_t)blockIdx.x * blockDim.x + threadIdx.x; i < total;
       i += (size_t)gridDim.x * blockDim.x) {
    if (i < nZero) {
      zero32[i] = 0u;
    } else if (i < nZero + nX) {
      size_t j = i - nZero;
      Xb[j] = __float2bfloat16(x[j]);
    } else {
      size_t j   = i - nZero - nX;              // linear over Wt (coalesced write)
      size_t col = j / ((size_t)G4 * K_SZ);
      size_t rem = j % ((size_t)G4 * K_SZ);
      size_t g   = rem / K_SZ;
      size_t k   = rem % K_SZ;
      Wt[j] = __float2bfloat16(Wg[g][k * H_SZ + col]);
    }
  }
}

// ---------------------------------------------------------------------------
// Persistent recurrent kernel: 32 WGs x 128 threads (4 waves)
//   WG (mwg, nwg): batch rows [32*mwg, +32), hidden cols [64*nwg, +64)
//   wave w: 16-col sub-slice, BOTH 16-row M-tiles, all 4 gates
// ---------------------------------------------------------------------------
__global__ void __launch_bounds__(128, 1)
lstm_persistent_kernel(const float* __restrict__ bc, const float* __restrict__ bi,
                       const float* __restrict__ bf, const float* __restrict__ bo,
                       float* __restrict__ out, unsigned char* __restrict__ ws) {
  constexpr int ZST  = K_SZ + 8;                 // LDS row stride (+16B pad)
  constexpr int MT1  = 16 * ZST;                 // elem offset of 2nd M-tile
  __shared__ unsigned short zsh[32 * ZST];       // z tile (32 x 1536), ~99 KB

  unsigned*             cnt  = (unsigned*)(ws + OFF_CNT);
  __hip_bfloat16*       hbuf = (__hip_bfloat16*)(ws + OFF_HBUF);
  const __hip_bfloat16* Xb   = (const __hip_bfloat16*)(ws + OFF_XB);
  const __hip_bfloat16* Wt   = (const __hip_bfloat16*)(ws + OFF_WT);

  const int mwg   = blockIdx.x & 1;              // batch tile 0..1 (32 rows)
  const int nwg   = blockIdx.x >> 1;             // hidden slice 0..15 (64 cols)
  const int wave  = threadIdx.x >> 5;
  const int lane  = threadIdx.x & 31;
  const int lcol  = lane & 15;
  const int kh8   = (lane >> 4) * 16;            // A-frag K sub-offset (elems)
  const int kh16  = (lane >> 4) * 16;            // B-frag K sub-offset (elems)
  const int bm0   = mwg * 32;
  const int rbase = (lane >> 4) * 8;             // D-frag row base within tile

  const int jcol = nwg * 64 + wave * 16 + lcol;  // this lane's hidden column
  const float bcv = bc[jcol], biv = bi[jcol], bfv = bf[jcol], bov = bo[jcol];

  // single B base: all 4 gates x whole K reachable via immediate offsets
  const __hip_bfloat16* wbase = Wt + (size_t)jcol * (G4 * K_SZ) + kh16;

  v8f creg[2];
#pragma unroll
  for (int mt = 0; mt < 2; ++mt)
#pragma unroll
    for (int v = 0; v < 8; ++v) creg[mt][v] = 0.0f;

  unsigned* mycnt = cnt + mwg;

  for (int t = 0; t < T_SZ; ++t) {
    const __hip_bfloat16* hb_cur = hbuf + (size_t)(t & 1) * B_SZ * H_SZ;
    __hip_bfloat16*       hb_nxt = hbuf + (size_t)((t + 1) & 1) * B_SZ * H_SZ;

    // ---- stage z = [x_t | h_{t-1}] (32 x 1536 bf16) into LDS --------------
    for (int q = threadIdx.x; q < 32 * 192; q += 128) {  // 192 16B-chunks/row
      int r = q / 192, c = q % 192;
      int b = bm0 + r;
      uint4 val;
      if (c < 64) {
        const __hip_bfloat16* xp = Xb + ((size_t)b * T_SZ + t) * D_SZ + c * 8;
        val = *(const uint4*)xp;
        if (t + 1 < T_SZ)
          __builtin_prefetch(xp + D_SZ, 0, 1);    // next step's x row chunk
      } else {
        val = *(const uint4*)(hb_cur + (size_t)b * H_SZ + (c - 64) * 8);
      }
      *(uint4*)(&zsh[r * ZST + c * 8]) = val;
    }
    __syncthreads();

    v8f acc[2][4];
#pragma unroll
    for (int mt = 0; mt < 2; ++mt)
#pragma unroll
      for (int g = 0; g < 4; ++g)
#pragma unroll
        for (int v = 0; v < 8; ++v) acc[mt][g][v] = 0.0f;

    // ---- K loop: 2 k-steps per iter; B 2-slot pipelined, A just-in-time ---
    // wptr/aptr advance once per iter; every load uses an immediate offset.
    const unsigned short*  aptr = &zsh[(lane & 15) * ZST + kh8];
    const __hip_bfloat16*  wptr = wbase;

    v16bf abuf[2];       // current k-step A frags (2 M-tiles)
    v16bf bbuf[2][4];    // 2-slot B pipeline x 4 gates

    auto loadA = [&](int off) {                   // off in elems: 0 or 32
      ((uint4*)&abuf[0])[0] = *(const uint4*)(aptr + off);
      ((uint4*)&abuf[0])[1] = *(const uint4*)(aptr + off + 8);
      ((uint4*)&abuf[1])[0] = *(const uint4*)(aptr + off + MT1);
      ((uint4*)&abuf[1])[1] = *(const uint4*)(aptr + off + MT1 + 8);
    };
    auto loadB = [&](int s, int off) {            // off in elems (K units)
#pragma unroll
      for (int g = 0; g < 4; ++g)
        bbuf[s][g] = *(const v16bf*)(wptr + g * K_SZ + off);
    };
    auto mmac = [&](int s) {
#pragma unroll
      for (int g = 0; g < 4; ++g) {
        acc[0][g] = __builtin_amdgcn_wmma_f32_16x16x32_bf16(
            false, abuf[0], false, bbuf[s][g], (short)0, acc[0][g],
            false, false);
        acc[1][g] = __builtin_amdgcn_wmma_f32_16x16x32_bf16(
            false, abuf[1], false, bbuf[s][g], (short)0, acc[1][g],
            false, false);
      }
    };

    loadB(0, 0);
    loadB(1, 32);
#pragma unroll 1
    for (int it = 0; it < (K_SZ / 64) - 1; ++it) {  // 23 steady iterations
      loadA(0);
      mmac(0);
      loadB(0, 128);            // refill slot 0: k-step +4, in flight 1 step
      loadA(32);
      mmac(1);
      loadB(1, 192);            // refill slot 1
      aptr += 64;
      wptr += 64;
    }
    loadA(0);
    mmac(0);                    // k-step K_SZ-64
    loadA(32);
    mmac(1);                    // k-step K_SZ-32

    // ---- lane-local LSTM cell update (D-frag: row = rbase+v, col = jcol) --
#pragma unroll
    for (int mt = 0; mt < 2; ++mt) {
#pragma unroll
      for (int v = 0; v < 8; ++v) {
        float cand = tanh_f(acc[mt][0][v] + bcv);
        float ig   = sigm_f(acc[mt][1][v] + biv);
        float fg   = sigm_f(acc[mt][2][v] + bfv);
        float cv   = fg * creg[mt][v] + ig * cand;
        float og   = sigm_f(acc[mt][3][v] + bov);
        float hv   = og * tanh_f(cv);
        creg[mt][v] = cv;
        int b = bm0 + mt * 16 + rbase + v;
        out[((size_t)b * T_SZ + t) * H_SZ + jcol] = hv;
        hb_nxt[(size_t)b * H_SZ + jcol] = __float2bfloat16(hv);
      }
    }

    // ---- per-batch-tile barrier (16 member WGs) ---------------------------
    __syncthreads();
    if (threadIdx.x == 0) {
      __threadfence();                                      // release h writes
      atomicAdd(mycnt, 1u);
      unsigned target = 16u * (unsigned)(t + 1);
      while (__hip_atomic_load(mycnt, __ATOMIC_RELAXED,
                               __HIP_MEMORY_SCOPE_AGENT) < target)
        __builtin_amdgcn_s_sleep(1);
      __threadfence();                                      // acquire
    }
    __syncthreads();
    __builtin_amdgcn_s_cluster_barrier();  // s_barrier -3; NOP w/o clusters
  }
}

// ---------------------------------------------------------------------------
extern "C" void kernel_launch(void* const* d_in, const int* in_sizes, int n_in,
                              void* d_out, int out_size, void* d_ws, size_t ws_size,
                              hipStream_t stream) {
  (void)in_sizes; (void)n_in; (void)out_size; (void)ws_size;
  const float* x  = (const float*)d_in[0];
  const float* Wc = (const float*)d_in[1];
  const float* bc = (const float*)d_in[2];
  const float* Wi = (const float*)d_in[3];
  const float* bi = (const float*)d_in[4];
  const float* Wf = (const float*)d_in[5];
  const float* bf = (const float*)d_in[6];
  const float* Wo = (const float*)d_in[7];
  const float* bo = (const float*)d_in[8];
  unsigned char* ws = (unsigned char*)d_ws;

  lstm_init_kernel<<<2048, 256, 0, stream>>>(x, Wc, Wi, Wf, Wo, ws);
  lstm_persistent_kernel<<<32, 128, 0, stream>>>(bc, bi, bf, bo,
                                                 (float*)d_out, ws);
}
